// Phaser_20452634264133
// MI455X (gfx1250) — matrix operand order repaired and checked by reading, not compiled, not run on hardware
//
#include <hip/hip_runtime.h>
#include <hip/hip_bf16.h>
#include <math.h>

// ---------------------------------------------------------------------------
// Phaser on MI455X (gfx1250, wave32).
//  * Per-hop MLP waveshaper runs on v_wmma_f32_16x16x32_f16 (16-row hop tiles).
//  * Both IIRs (biquad + time-varying 4th-order LPC) are parallelized as
//    chunked affine scans over 4x4 companion-state transition matrices.
// ---------------------------------------------------------------------------

typedef __attribute__((ext_vector_type(16))) _Float16 v16h;
typedef __attribute__((ext_vector_type(8)))  float    v8f;

namespace {
constexpr int NS     = 1 << 20;            // 1,048,576 samples
constexpr int HOPSZ  = 551;
constexpr int NHOPS  = NS / HOPSZ + 1;     // 1904 = 16 * 119
constexpr int MLPW   = 16;
constexpr int NTILES = NHOPS / 16;         // 119 wmma tiles
constexpr int CHUNKS = 8192;               // recurrence chunks
constexpr int CLEN   = NS / CHUNKS;        // 128 samples/chunk
constexpr int SCAN_T = 128;                // threads in scan block
constexpr int GROUP  = CHUNKS / SCAN_T;    // 64 chunks/thread in scan
}

// linear interp of a (NHOPS,5) coeff table at sample n (align_corners=True)
__device__ __forceinline__ void interp_row(const float* __restrict__ c, int n,
                                           float o[5]) {
  float pos = (float)n * ((float)(NHOPS - 1) / (float)(NS - 1));
  int i0 = (int)pos;
  if (i0 > NHOPS - 2) i0 = NHOPS - 2;
  float fr = pos - (float)i0;
  const float* r0 = c + i0 * 5;
#pragma unroll
  for (int k = 0; k < 5; ++k) o[k] = fmaf(fr, r0[5 + k] - r0[k], r0[k]);
}

// ---------------------------------------------------------------------------
// Kernel 1: LFO + MLP (WMMA) + allpass coefficient tables (cb_h, cd_h) + p.
// One wave per 16-hop tile. f16 WMMA layouts per ISA 7.12.2:
//   A (16x32 f16): lane = row(M)&15 + 16*half ; elem e -> K = (e<8?e:e+8)+8*half
//   B (32x16 f16): lane = col(N)&15 + 16*half ; elem e -> K = (e<8?e:e+8)+8*half
//   C/D (16x16 f32): lane l holds col=l&15, VGPR d holds row d + 8*(l>>4)
// ---------------------------------------------------------------------------
__global__ void __launch_bounds__(32) hops_kernel(
    const float* __restrict__ g2,     const float* __restrict__ depth,
    const float* __restrict__ biasp,  const float* __restrict__ omega,
    const float* __restrict__ phi,    const float* __restrict__ rlogit,
    const float* __restrict__ W_in,   const float* __restrict__ b_in,
    const float* __restrict__ W_h,    const float* __restrict__ b_h,
    const float* __restrict__ W_out,  const float* __restrict__ b_out,
    float* __restrict__ cb_h, float* __restrict__ cd_h,
    float* __restrict__ p_out) {
  __shared__ float lds0[16 * 16];
  __shared__ float lds1[16 * 16];

  const int lane = threadIdx.x;
  const int m    = lane & 15;   // A row / B,C,D column
  const int half = lane >> 4;
  const int col  = m;

  // damped-oscillator LFO, row m of this tile
  const float t   = (float)(blockIdx.x * 16 + m);
  const float r   = 1.0f / (1.0f + __expf(-rlogit[0]));
  const float amp = __expf(t * __logf(r));
  const float ph  = omega[0] * t + phi[0];
  const float l0  = amp * __cosf(ph);
  const float l1  = amp * __sinf(ph);

  v16h A, B;
  const v8f Cz = {};

  // ---- input layer: (16x2 pad 32) @ (2x16 pad 32x16) ----
#pragma unroll
  for (int e = 0; e < 16; ++e) {
    int Ka = ((e < 8) ? e : e + 8) + 8 * half;
    float av = (Ka == 0) ? l0 : (Ka == 1) ? l1 : 0.0f;
    A[e] = (_Float16)av;
    int Kb = ((e < 8) ? e : e + 8) + 8 * half;
    B[e] = (_Float16)((Kb < 2) ? W_in[Kb * MLPW + col] : 0.0f);
  }
  v8f C = __builtin_amdgcn_wmma_f32_16x16x32_f16(false, A, false, B,
                                                 (short)0, Cz, false, false);
#pragma unroll
  for (int d = 0; d < 8; ++d)
    lds0[(d + 8 * half) * 16 + col] = tanhf(C[d] + b_in[col]);

  // ---- two hidden 16x16 layers, LDS ping-pong (single wave: DS in-order) ----
  float* src = lds0;
  float* dst = lds1;
#pragma unroll
  for (int layer = 0; layer < 2; ++layer) {
#pragma unroll
    for (int e = 0; e < 16; ++e) {
      int Ka = ((e < 8) ? e : e + 8) + 8 * half;
      A[e] = (_Float16)((Ka < 16) ? src[m * 16 + Ka] : 0.0f);
      int Kb = ((e < 8) ? e : e + 8) + 8 * half;
      B[e] = (_Float16)((Kb < 16) ? W_h[layer * 256 + Kb * 16 + col] : 0.0f);
    }
    C = __builtin_amdgcn_wmma_f32_16x16x32_f16(false, A, false, B,
                                               (short)0, Cz, false, false);
#pragma unroll
    for (int d = 0; d < 8; ++d)
      dst[(d + 8 * half) * 16 + col] = tanhf(C[d] + b_h[layer * 16 + col]);
    float* tmp = src; src = dst; dst = tmp;
  }

  // ---- output layer (16->1) + allpass coefficient tables ----
  if (half == 0) {
    float acc = b_out[0];
#pragma unroll
    for (int k = 0; k < 16; ++k) acc = fmaf(src[m * 16 + k], W_out[k], acc);
    float wsv = tanhf(acc);
    float dv  = biasp[0] + depth[0] * 0.5f * (1.0f + wsv);
    float td  = tanf(dv);
    float pv  = tanhf((1.0f - td) / (1.0f + td));
    int hop = blockIdx.x * 16 + m;
    if (hop < NHOPS) {
      float p2 = pv * pv, p3 = p2 * pv, p4 = p2 * p2;
      float bap[5] = {p4, -4.f * p3, 6.f * p2, -4.f * pv, 1.f};
      float aap[5] = {1.f, -4.f * pv, 6.f * p2, -4.f * p3, p4};
      float g2a = fabsf(g2[0]);
      float inv = 1.0f / (aap[0] - g2a * bap[0]);
#pragma unroll
      for (int k = 0; k < 5; ++k) {
        float dk = aap[k] - g2a * bap[k];
        cb_h[hop * 5 + k] = bap[k] * inv;
        cd_h[hop * 5 + k] = dk * inv;
      }
      p_out[hop] = pv;
    }
  }
}

// ---------------------------------------------------------------------------
// Chunked linear-recurrence machinery. State s[n] = (y[n-1..n-4]).
// MODE 0: biquad (constant 3-tap FIR + order-2 IIR padded to order 4).
// MODE 1: time-varying 5-tap FIR (cb) + order-4 IIR (cd), interp per sample.
// ---------------------------------------------------------------------------
template <int MODE>
__device__ __forceinline__ void sample_va(const float* __restrict__ sig, int n,
                                          const float* __restrict__ cb_h,
                                          const float* __restrict__ cd_h,
                                          float f0, float f1, float f2,
                                          float a1c, float a2c,
                                          float& v, float a[4]) {
  if (MODE == 0) {
    a[0] = a1c; a[1] = a2c; a[2] = 0.f; a[3] = 0.f;
    float x0  = sig[n];
    float xm1 = (n >= 1) ? sig[n - 1] : 0.f;
    float xm2 = (n >= 2) ? sig[n - 2] : 0.f;
    v = fmaf(f2, xm2, fmaf(f1, xm1, f0 * x0));
  } else {
    float cbv[5], cdv[5];
    interp_row(cb_h, n, cbv);
    interp_row(cd_h, n, cdv);
    a[0] = cdv[1]; a[1] = cdv[2]; a[2] = cdv[3]; a[3] = cdv[4];
    v = cbv[0] * sig[n];
#pragma unroll
    for (int k = 1; k < 5; ++k) v = fmaf(cbv[k], (n >= k) ? sig[n - k] : 0.f, v);
  }
}

template <int MODE>
__global__ void __launch_bounds__(256) chunk_reduce(
    const float* __restrict__ sig,
    const float* __restrict__ cb_h, const float* __restrict__ cd_h,
    const float* __restrict__ bq_dc, const float* __restrict__ bq_ff,
    const float* __restrict__ bq_fb,
    float* __restrict__ Tmat, float* __restrict__ dvec) {
  int c = blockIdx.x * blockDim.x + threadIdx.x;
  if (c >= CHUNKS) return;
  float a1c = 0.f, a2c = 0.f, f0 = 0.f, f1 = 0.f, f2 = 0.f;
  if (MODE == 0) {
    float A1 = 2.f * tanhf(bq_fb[0]);
    a1c = A1;
    a2c = 0.5f * ((2.f - fabsf(A1)) * tanhf(bq_fb[1]) + fabsf(A1));
    f0 = bq_dc[0]; f1 = bq_ff[0]; f2 = bq_ff[1];
  }
  float z0 = 0, z1 = 0, z2 = 0, z3 = 0;                 // zero-state
  float S[4][4] = {{1, 0, 0, 0}, {0, 1, 0, 0}, {0, 0, 1, 0}, {0, 0, 0, 1}};
  const int base = c * CLEN;
  for (int i = 0; i < CLEN; ++i) {
    int n = base + i;
    float v, a[4];
    sample_va<MODE>(sig, n, cb_h, cd_h, f0, f1, f2, a1c, a2c, v, a);
    float y = v - a[0] * z0 - a[1] * z1 - a[2] * z2 - a[3] * z3;
    z3 = z2; z2 = z1; z1 = z0; z0 = y;
#pragma unroll
    for (int j = 0; j < 4; ++j) {
      float h = -(a[0] * S[0][j] + a[1] * S[1][j] + a[2] * S[2][j] + a[3] * S[3][j]);
      S[3][j] = S[2][j]; S[2][j] = S[1][j]; S[1][j] = S[0][j]; S[0][j] = h;
    }
  }
#pragma unroll
  for (int k = 0; k < 4; ++k)
#pragma unroll
    for (int j = 0; j < 4; ++j) Tmat[c * 16 + k * 4 + j] = S[k][j];
  dvec[c * 4 + 0] = z0; dvec[c * 4 + 1] = z1;
  dvec[c * 4 + 2] = z2; dvec[c * 4 + 3] = z3;
}

// single-block hierarchical scan of affine maps s' = T s + d -> per-chunk init
__global__ void __launch_bounds__(SCAN_T) affine_scan(
    const float* __restrict__ Tmat, const float* __restrict__ dvec,
    float* __restrict__ sinit) {
  __shared__ float gT[SCAN_T * 16];
  __shared__ float gd[SCAN_T * 4];
  __shared__ float pre[SCAN_T * 4];
  const int t = threadIdx.x;

  // phase 1: compose this thread's GROUP consecutive chunk maps
  float Tc[16] = {1, 0, 0, 0, 0, 1, 0, 0, 0, 0, 1, 0, 0, 0, 0, 1};
  float dc[4]  = {0, 0, 0, 0};
  for (int i = 0; i < GROUP; ++i) {
    int c = t * GROUP + i;
    const float* Tn = Tmat + c * 16;
    const float* dn = dvec + c * 4;
    float R[16], e[4];
#pragma unroll
    for (int r2 = 0; r2 < 4; ++r2) {
#pragma unroll
      for (int q = 0; q < 4; ++q) {
        float s = 0.f;
#pragma unroll
        for (int k = 0; k < 4; ++k) s = fmaf(Tn[r2 * 4 + k], Tc[k * 4 + q], s);
        R[r2 * 4 + q] = s;
      }
      float sv = dn[r2];
#pragma unroll
      for (int k = 0; k < 4; ++k) sv = fmaf(Tn[r2 * 4 + k], dc[k], sv);
      e[r2] = sv;
    }
#pragma unroll
    for (int k = 0; k < 16; ++k) Tc[k] = R[k];
#pragma unroll
    for (int k = 0; k < 4; ++k) dc[k] = e[k];
  }
#pragma unroll
  for (int k = 0; k < 16; ++k) gT[t * 16 + k] = Tc[k];
#pragma unroll
  for (int k = 0; k < 4; ++k) gd[t * 4 + k] = dc[k];
  __syncthreads();

  // phase 2: serial scan over SCAN_T group aggregates
  if (t == 0) {
    float s[4] = {0, 0, 0, 0};
    for (int g = 0; g < SCAN_T; ++g) {
#pragma unroll
      for (int k = 0; k < 4; ++k) pre[g * 4 + k] = s[k];
      float ns[4];
#pragma unroll
      for (int r2 = 0; r2 < 4; ++r2) {
        ns[r2] = gd[g * 4 + r2];
#pragma unroll
        for (int k = 0; k < 4; ++k)
          ns[r2] = fmaf(gT[g * 16 + r2 * 4 + k], s[k], ns[r2]);
      }
#pragma unroll
      for (int k = 0; k < 4; ++k) s[k] = ns[k];
    }
  }
  __syncthreads();

  // phase 3: replay inside each group, emitting per-chunk initial states
  float s[4];
#pragma unroll
  for (int k = 0; k < 4; ++k) s[k] = pre[t * 4 + k];
  for (int i = 0; i < GROUP; ++i) {
    int c = t * GROUP + i;
#pragma unroll
    for (int k = 0; k < 4; ++k) sinit[c * 4 + k] = s[k];
    const float* Tn = Tmat + c * 16;
    const float* dn = dvec + c * 4;
    float ns[4];
#pragma unroll
    for (int r2 = 0; r2 < 4; ++r2) {
      ns[r2] = dn[r2];
#pragma unroll
      for (int k = 0; k < 4; ++k) ns[r2] = fmaf(Tn[r2 * 4 + k], s[k], ns[r2]);
    }
#pragma unroll
    for (int k = 0; k < 4; ++k) s[k] = ns[k];
  }
}

template <int MODE>
__global__ void __launch_bounds__(256) chunk_apply(
    const float* __restrict__ sig,
    const float* __restrict__ cb_h, const float* __restrict__ cd_h,
    const float* __restrict__ bq_dc, const float* __restrict__ bq_ff,
    const float* __restrict__ bq_fb,
    const float* __restrict__ sinit, const float* __restrict__ g1,
    float* __restrict__ out) {
  int c = blockIdx.x * blockDim.x + threadIdx.x;
  if (c >= CHUNKS) return;
  float a1c = 0.f, a2c = 0.f, f0 = 0.f, f1 = 0.f, f2 = 0.f;
  if (MODE == 0) {
    float A1 = 2.f * tanhf(bq_fb[0]);
    a1c = A1;
    a2c = 0.5f * ((2.f - fabsf(A1)) * tanhf(bq_fb[1]) + fabsf(A1));
    f0 = bq_dc[0]; f1 = bq_ff[0]; f2 = bq_ff[1];
  }
  const float g1v = (MODE == 1) ? g1[0] : 0.f;
  float z0 = sinit[c * 4 + 0], z1 = sinit[c * 4 + 1];
  float z2 = sinit[c * 4 + 2], z3 = sinit[c * 4 + 3];
  const int base = c * CLEN;
  for (int i = 0; i < CLEN; ++i) {
    int n = base + i;
    float v, a[4];
    sample_va<MODE>(sig, n, cb_h, cd_h, f0, f1, f2, a1c, a2c, v, a);
    float y = v - a[0] * z0 - a[1] * z1 - a[2] * z2 - a[3] * z3;
    z3 = z2; z2 = z1; z1 = z0; z0 = y;
    out[n] = (MODE == 0) ? y : fmaf(g1v, sig[n], y);
  }
}

// ---------------------------------------------------------------------------
// Launch: hops(WMMA) -> biquad reduce/scan/apply(h1) -> LPC reduce/scan/apply.
// Workspace (~4.83 MB floats): cb_h | cd_h | h1 | T | d | sinit
// ---------------------------------------------------------------------------
extern "C" void kernel_launch(void* const* d_in, const int* in_sizes, int n_in,
                              void* d_out, int out_size, void* d_ws, size_t ws_size,
                              hipStream_t stream) {
  const float* x       = (const float*)d_in[0];
  const float* g1      = (const float*)d_in[1];
  const float* g2      = (const float*)d_in[2];
  const float* depth   = (const float*)d_in[3];
  const float* biasp   = (const float*)d_in[4];
  const float* omega   = (const float*)d_in[5];
  const float* phi     = (const float*)d_in[6];
  const float* rlogit  = (const float*)d_in[7];
  const float* W_in    = (const float*)d_in[8];
  const float* b_in    = (const float*)d_in[9];
  const float* W_h     = (const float*)d_in[10];
  const float* b_h     = (const float*)d_in[11];
  const float* W_out   = (const float*)d_in[12];
  const float* b_out   = (const float*)d_in[13];
  const float* bq_dc   = (const float*)d_in[14];
  const float* bq_ff   = (const float*)d_in[15];
  const float* bq_fb   = (const float*)d_in[16];

  float* out   = (float*)d_out;          // [0,NS): signal, [NS,NS+NHOPS): p
  float* ws    = (float*)d_ws;
  float* cb_h  = ws;                     // NHOPS*5
  float* cd_h  = cb_h + NHOPS * 5;       // NHOPS*5
  float* h1    = cd_h + NHOPS * 5;       // NS
  float* Tm    = h1 + NS;                // CHUNKS*16
  float* dv    = Tm + CHUNKS * 16;       // CHUNKS*4
  float* si    = dv + CHUNKS * 4;        // CHUNKS*4

  // 1) per-hop coefficients via WMMA MLP
  hipLaunchKernelGGL(hops_kernel, dim3(NTILES), dim3(32), 0, stream,
                     g2, depth, biasp, omega, phi, rlogit,
                     W_in, b_in, W_h, b_h, W_out, b_out,
                     cb_h, cd_h, out + NS);

  const int cblocks = CHUNKS / 256;
  // 2) biquad: h1 = lfilter(x)
  hipLaunchKernelGGL((chunk_reduce<0>), dim3(cblocks), dim3(256), 0, stream,
                     x, cb_h, cd_h, bq_dc, bq_ff, bq_fb, Tm, dv);
  hipLaunchKernelGGL(affine_scan, dim3(1), dim3(SCAN_T), 0, stream, Tm, dv, si);
  hipLaunchKernelGGL((chunk_apply<0>), dim3(cblocks), dim3(256), 0, stream,
                     x, cb_h, cd_h, bq_dc, bq_ff, bq_fb, si, g1, h1);

  // 3) time-varying FIR + LPC on h1; out = g1*h1 + y2
  hipLaunchKernelGGL((chunk_reduce<1>), dim3(cblocks), dim3(256), 0, stream,
                     h1, cb_h, cd_h, bq_dc, bq_ff, bq_fb, Tm, dv);
  hipLaunchKernelGGL(affine_scan, dim3(1), dim3(SCAN_T), 0, stream, Tm, dv, si);
  hipLaunchKernelGGL((chunk_apply<1>), dim3(cblocks), dim3(256), 0, stream,
                     h1, cb_h, cd_h, bq_dc, bq_ff, bq_fb, si, g1, out);
}
